// re2graph_onlygat_10754598109603
// MI455X (gfx1250) — compile-verified, matrix operand order checked by reading, not run on hardware
//
#include <hip/hip_runtime.h>
#include <stdint.h>

#define ALPHA 0.2f
#define NEG_BIG -9.0e15f

typedef __attribute__((ext_vector_type(16))) __bf16 v16bf;
typedef __attribute__((ext_vector_type(8)))  float  v8f;

#if defined(__HIP_DEVICE_COMPILE__) && __has_builtin(__builtin_amdgcn_tensor_load_to_lds)
#define HAVE_TDM 1
#else
#define HAVE_TDM 0
#endif

__device__ __forceinline__ uint16_t f2bf(float f) {
  union { float f; uint32_t u; } c; c.f = f;
  uint32_t u = c.u;
  uint32_t r = u + 0x7FFFu + ((u >> 16) & 1u);   // round-to-nearest-even
  return (uint16_t)(r >> 16);
}
__device__ __forceinline__ float bf2f(uint16_t b) {
  union { uint32_t u; float f; } c; c.u = ((uint32_t)b) << 16; return c.f;
}

#if HAVE_TDM
typedef uint32_t tdm_v4u __attribute__((ext_vector_type(4)));
typedef int      tdm_v8i __attribute__((ext_vector_type(8)));
typedef int      tdm_v4i __attribute__((ext_vector_type(4)));

// Issue a TDM 2D tile load: tile_dim0=32 elems (bf16), tile_dim1=rows,
// tensor row pitch = 1024 elems.  D# packing per CDNA5 ISA §8.3/8.4.
__device__ __forceinline__ void tdm_load_tile(uint32_t lds_off, const void* gsrc,
                                              uint32_t rows, uint32_t tensor_rows) {
  uint64_t ga = (uint64_t)(uintptr_t)gsrc;
  tdm_v4u g0;
  g0[0] = 1u;                                          // count=1, user descriptor
  g0[1] = lds_off;                                     // lds_addr (bytes)
  g0[2] = (uint32_t)ga;                                // global_addr[31:0]
  g0[3] = (uint32_t)((ga >> 32) & 0x01FFFFFFu) | (2u << 30);  // addr[56:32], type=2
  tdm_v8i g1;
  g1[0] = (int)(1u << 16);                             // wg_mask=0, data_size=1 (2B)
  g1[1] = (int)(1024u << 16);                          // tensor_dim0[15:0]=1024
  g1[2] = (int)((tensor_rows & 0xFFFFu) << 16);        // tensor_dim1[15:0]
  g1[3] = (int)((tensor_rows >> 16) & 0xFFFFu) | (int)(32u << 16); // dim1 hi | tile_dim0=32
  g1[4] = (int)(rows & 0xFFFFu);                       // tile_dim1 | tile_dim2=0
  g1[5] = 1024;                                        // tensor_dim0_stride[31:0]
  g1[6] = 0;                                           // stride hi | dim1_stride lo
  g1[7] = 0;
  tdm_v4i gz = {0, 0, 0, 0};
#if __clang_major__ >= 23
  tdm_v8i gz8 = {0, 0, 0, 0, 0, 0, 0, 0};
  __builtin_amdgcn_tensor_load_to_lds(g0, g1, gz, gz, gz8, 0);
#else
  __builtin_amdgcn_tensor_load_to_lds(g0, g1, gz, gz, 0);
#endif
}
#endif

// ---------------------------------------------------------------------------
// 1) fp32 -> bf16 bulk convert (4 elements / thread)
// ---------------------------------------------------------------------------
__global__ __launch_bounds__(256) void cvt_f32_bf16_kernel(
    const float* __restrict__ src, uint16_t* __restrict__ dst, int n4) {
  int i = blockIdx.x * 256 + threadIdx.x;
  if (i < n4) {
    float4 f = ((const float4*)src)[i];
    uint2 p;
    p.x = (uint32_t)f2bf(f.x) | ((uint32_t)f2bf(f.y) << 16);
    p.y = (uint32_t)f2bf(f.z) | ((uint32_t)f2bf(f.w) << 16);
    ((uint2*)dst)[i] = p;
  }
}

// ---------------------------------------------------------------------------
// 2) W (fp32 [k][n]) -> WT (bf16 [n][k])
// ---------------------------------------------------------------------------
__global__ __launch_bounds__(256) void wt_kernel(
    const float* __restrict__ W, uint16_t* __restrict__ WT) {
  int idx = blockIdx.x * 256 + threadIdx.x;   // 1M
  int n = idx >> 10, k = idx & 1023;
  WT[idx] = f2bf(W[k * 1024 + n]);            // WT[n*1024+k]
}

// ---------------------------------------------------------------------------
// 3) w1[k] = dot(W[k,:], a[:D]), w2[k] = dot(W[k,:], a[D:])
// ---------------------------------------------------------------------------
__global__ __launch_bounds__(256) void wvec_kernel(
    const float* __restrict__ W, const float* __restrict__ a,
    float* __restrict__ w1, float* __restrict__ w2) {
  __shared__ float r1[256], r2[256];
  int k = blockIdx.x, t = threadIdx.x;
  float s1 = 0.f, s2 = 0.f;
  for (int n = t; n < 1024; n += 256) {
    float wv = W[k * 1024 + n];
    s1 += wv * a[n];
    s2 += wv * a[1024 + n];
  }
  r1[t] = s1; r2[t] = s2; __syncthreads();
  for (int s = 128; s > 0; s >>= 1) {
    if (t < s) { r1[t] += r1[t + s]; r2[t] += r2[t + s]; }
    __syncthreads();
  }
  if (t == 0) { w1[k] = r1[0]; w2[k] = r2[0]; }
}

// ---------------------------------------------------------------------------
// 4) Wh1[i] = dot(h[i,:], w1), Wh2[i] = dot(h[i,:], w2)   (h in bf16)
// ---------------------------------------------------------------------------
__global__ __launch_bounds__(256) void wh12_kernel(
    const uint16_t* __restrict__ hb,
    const float* __restrict__ w1, const float* __restrict__ w2,
    float* __restrict__ Wh1, float* __restrict__ Wh2) {
  __shared__ float r1[256], r2[256];
  int i = blockIdx.x, t = threadIdx.x;
  const uint16_t* hrow = hb + (size_t)i * 1024;
  float s1 = 0.f, s2 = 0.f;
  for (int k = t; k < 1024; k += 256) {
    float v = bf2f(hrow[k]);
    s1 += v * w1[k];
    s2 += v * w2[k];
  }
  r1[t] = s1; r2[t] = s2; __syncthreads();
  for (int s = 128; s > 0; s >>= 1) {
    if (t < s) { r1[t] += r1[t + s]; r2[t] += r2[t + s]; }
    __syncthreads();
  }
  if (t == 0) { Wh1[i] = r1[0]; Wh2[i] = r2[0]; }
}

// ---------------------------------------------------------------------------
// 5) GEMM1: Wh = hb(bf16 [32768][1024]) @ W  via WT(bf16 [n][k]).
//    Output stored transposed per batch: WhT[b][n][i] (bf16).
//    128x128 block tile, 8 waves (each 32x64 = 2x4 WMMA tiles), K-step 32.
//    TDM double-buffered LDS staging (fallback: vector loads).
// ---------------------------------------------------------------------------
__global__ __launch_bounds__(256) void gemm1_bf16_kernel(
    const uint16_t* __restrict__ A,   // [M][1024] bf16 (h)
    const uint16_t* __restrict__ BT,  // [1024][1024] bf16 (W^T, [n][k])
    uint16_t* __restrict__ CT) {      // [b][n][i] bf16 (Wh^T)
  __shared__ __align__(16) uint16_t As[2][128 * 32];
  __shared__ __align__(16) uint16_t Bs[2][128 * 32];

  const int n0 = blockIdx.x * 128;
  const int m0 = blockIdx.y * 128;
  const int t = threadIdx.x;
  const int lane = t & 31;
  const int wave = t >> 5;
  const int wm = (wave & 3) * 32;
  const int wn = (wave >> 2) * 64;
  const int lrow = lane & 15;
  const int lhalf = lane >> 4;

  const v8f vzero = {0.f, 0.f, 0.f, 0.f, 0.f, 0.f, 0.f, 0.f};
  v8f acc[2][4];
#pragma unroll
  for (int i = 0; i < 2; i++)
#pragma unroll
    for (int j = 0; j < 4; j++) acc[i][j] = vzero;

  union Frag { v16bf v; uint4 u[2]; };

#if HAVE_TDM
  const bool issuer = (wave == 0);
  const uint32_t asBase = (uint32_t)(size_t)&As[0][0];
  const uint32_t bsBase = (uint32_t)(size_t)&Bs[0][0];
  if (issuer) {
    tdm_load_tile(asBase, A + (size_t)m0 * 1024, 128, 32768);
    tdm_load_tile(bsBase, BT + (size_t)n0 * 1024, 128, 1024);
  }
  for (int k0 = 0; k0 < 1024; k0 += 32) {
    const int cur = (k0 >> 5) & 1;
    const int nxt = cur ^ 1;
    if (issuer) {
      if (k0 + 32 < 1024) {   // prefetch next stage, wait for current
        tdm_load_tile(asBase + (uint32_t)nxt * 8192u,
                      A + (size_t)m0 * 1024 + k0 + 32, 128, 32768);
        tdm_load_tile(bsBase + (uint32_t)nxt * 8192u,
                      BT + (size_t)n0 * 1024 + k0 + 32, 128, 1024);
        __builtin_amdgcn_s_wait_tensorcnt(2);
      } else {
        __builtin_amdgcn_s_wait_tensorcnt(0);
      }
    }
    __syncthreads();
    const uint16_t* Ap = &As[cur][0];
    const uint16_t* Bq = &Bs[cur][0];
#else
  const int row = t >> 1;
  const int seg = (t & 1) * 16;     // 16 bf16 = 32B per thread per tile
  for (int k0 = 0; k0 < 1024; k0 += 32) {
    __syncthreads();
    {
      const uint16_t* gp = A + (size_t)(m0 + row) * 1024 + k0 + seg;
      uint4 d0 = *(const uint4*)(gp);
      uint4 d1 = *(const uint4*)(gp + 8);
      *(uint4*)&As[0][row * 32 + seg] = d0;
      *(uint4*)&As[0][row * 32 + seg + 8] = d1;
    }
    {
      const uint16_t* gp = BT + (size_t)(n0 + row) * 1024 + k0 + seg;
      uint4 d0 = *(const uint4*)(gp);
      uint4 d1 = *(const uint4*)(gp + 8);
      *(uint4*)&Bs[0][row * 32 + seg] = d0;
      *(uint4*)&Bs[0][row * 32 + seg + 8] = d1;
    }
    __syncthreads();
    const uint16_t* Ap = &As[0][0];
    const uint16_t* Bq = &Bs[0][0];
#endif

    Frag af[2], bfr[4];
#pragma unroll
    for (int mi = 0; mi < 2; mi++) {
      int r = wm + mi * 16 + lrow;
      af[mi].u[0] = *(const uint4*)&Ap[r * 32 + lhalf * 8];
      af[mi].u[1] = *(const uint4*)&Ap[r * 32 + 16 + lhalf * 8];
    }
#pragma unroll
    for (int ni = 0; ni < 4; ni++) {
      int r = wn + ni * 16 + lrow;
      bfr[ni].u[0] = *(const uint4*)&Bq[r * 32 + lhalf * 8];
      bfr[ni].u[1] = *(const uint4*)&Bq[r * 32 + 16 + lhalf * 8];
    }
#pragma unroll
    for (int mi = 0; mi < 2; mi++)
#pragma unroll
      for (int ni = 0; ni < 4; ni++)
        acc[mi][ni] = __builtin_amdgcn_wmma_f32_16x16x32_bf16(
            false, af[mi].v, false, bfr[ni].v, (short)0, acc[mi][ni], false, false);
#if HAVE_TDM
    __syncthreads();
#endif
  }

  // epilogue: lane holds fixed n = lrow, m = v + 8*lhalf (+tile offsets);
  // 8 consecutive m -> one b128 store into WhT[b][n][i]
#pragma unroll
  for (int mi = 0; mi < 2; mi++)
#pragma unroll
    for (int ni = 0; ni < 4; ni++) {
      int m_local = wm + mi * 16 + 8 * lhalf;
      int n = n0 + wn + ni * 16 + lrow;
      int m = m0 + m_local;
      int bidx = m >> 10;
      int i0 = m & 1023;
      __align__(16) uint16_t h8[8];
#pragma unroll
      for (int v = 0; v < 8; v++) h8[v] = f2bf(acc[mi][ni][v]);
      *(uint4*)&CT[(size_t)bidx * 1048576 + (size_t)n * 1024 + i0] =
          *(const uint4*)h8;
    }
}

// ---------------------------------------------------------------------------
// 6) softmax: att[b,i,j] = softmax_j( mask_j ? leaky(Wh1_i + Wh2_j) : NEG )
// ---------------------------------------------------------------------------
__global__ __launch_bounds__(256) void softmax_att_kernel(
    const float* __restrict__ Wh1, const float* __restrict__ Wh2,
    const int* __restrict__ mask, float* __restrict__ att) {
  __shared__ float red[256];
  const int rowg = blockIdx.x;          // b*S + i
  const int b = rowg >> 10;
  const int t = threadIdx.x;
  const float s1 = Wh1[rowg];
  float e[4];
  float mx = -3.0e38f;
#pragma unroll
  for (int q = 0; q < 4; q++) {
    int j = t + q * 256;
    float v = s1 + Wh2[(b << 10) + j];
    v = v > 0.0f ? v : ALPHA * v;
    if (mask[(b << 10) + j] <= 0) v = NEG_BIG;
    e[q] = v;
    mx = fmaxf(mx, v);
  }
  red[t] = mx; __syncthreads();
  for (int s = 128; s > 0; s >>= 1) {
    if (t < s) red[t] = fmaxf(red[t], red[t + s]);
    __syncthreads();
  }
  mx = red[0]; __syncthreads();
  float sum = 0.f;
#pragma unroll
  for (int q = 0; q < 4; q++) { e[q] = __expf(e[q] - mx); sum += e[q]; }
  red[t] = sum; __syncthreads();
  for (int s = 128; s > 0; s >>= 1) {
    if (t < s) red[t] += red[t + s];
    __syncthreads();
  }
  float inv = 1.0f / red[0];
#pragma unroll
  for (int q = 0; q < 4; q++)
    att[(size_t)rowg * 1024 + t + q * 256] = e[q] * inv;
}

// ---------------------------------------------------------------------------
// 7) GEMM2: out = att(fp32, cvt->bf16 in LDS) @ Wh (via WhT bf16) + h
//    B tile via TDM double-buffer; A tile manual (needs fp32->bf16 cvt).
// ---------------------------------------------------------------------------
__global__ __launch_bounds__(256) void gemm2_bf16_kernel(
    const float* __restrict__ Aatt,     // [B*S][1024] fp32 (att)
    const uint16_t* __restrict__ WhT,   // [b][n][k] bf16
    const float* __restrict__ resid,    // [B*S][1024] fp32 (h)
    float* __restrict__ out) {          // [B*S][1024] fp32
  __shared__ __align__(16) uint16_t As[128 * 32];
  __shared__ __align__(16) uint16_t Bs[2][128 * 32];

  const int n0 = blockIdx.x * 128;
  const int m0 = blockIdx.y * 128;
  const int t = threadIdx.x;
  const int lane = t & 31;
  const int wave = t >> 5;
  const int wm = (wave & 3) * 32;
  const int wn = (wave >> 2) * 64;
  const int lrow = lane & 15;
  const int lhalf = lane >> 4;
  const uint16_t* Bp = WhT + ((size_t)(m0 >> 10)) * 1048576;  // batch base

  const v8f vzero = {0.f, 0.f, 0.f, 0.f, 0.f, 0.f, 0.f, 0.f};
  v8f acc[2][4];
#pragma unroll
  for (int i = 0; i < 2; i++)
#pragma unroll
    for (int j = 0; j < 4; j++) acc[i][j] = vzero;

  const int row = t >> 1;
  const int seg = (t & 1) * 16;

  union Frag { v16bf v; uint4 u[2]; };

#if HAVE_TDM
  const bool issuer = (wave == 0);
  const uint32_t bsBase = (uint32_t)(size_t)&Bs[0][0];
  if (issuer) tdm_load_tile(bsBase, Bp + (size_t)n0 * 1024, 128, 1024);
#endif

  for (int k0 = 0; k0 < 1024; k0 += 32) {
    __syncthreads();
    {   // A tile: 16 fp32 per thread -> 16 bf16
      const float* gp = Aatt + (size_t)(m0 + row) * 1024 + k0 + seg;
      __align__(16) uint16_t cv[16];
#pragma unroll
      for (int q = 0; q < 4; q++) {
        float4 f = ((const float4*)gp)[q];
        cv[q * 4 + 0] = f2bf(f.x); cv[q * 4 + 1] = f2bf(f.y);
        cv[q * 4 + 2] = f2bf(f.z); cv[q * 4 + 3] = f2bf(f.w);
      }
      *(uint4*)&As[row * 32 + seg] = *(const uint4*)&cv[0];
      *(uint4*)&As[row * 32 + seg + 8] = *(const uint4*)&cv[8];
    }
#if HAVE_TDM
    const int cur = (k0 >> 5) & 1;
    const int nxt = cur ^ 1;
    if (issuer) {
      if (k0 + 32 < 1024) {
        tdm_load_tile(bsBase + (uint32_t)nxt * 8192u,
                      Bp + (size_t)n0 * 1024 + k0 + 32, 128, 1024);
        __builtin_amdgcn_s_wait_tensorcnt(1);
      } else {
        __builtin_amdgcn_s_wait_tensorcnt(0);
      }
    }
    const uint16_t* Bq = &Bs[cur][0];
#else
    {   // B tile: WhT rows (n, k contiguous)
      const uint16_t* gp = Bp + (size_t)(n0 + row) * 1024 + k0 + seg;
      uint4 d0 = *(const uint4*)(gp);
      uint4 d1 = *(const uint4*)(gp + 8);
      *(uint4*)&Bs[0][row * 32 + seg] = d0;
      *(uint4*)&Bs[0][row * 32 + seg + 8] = d1;
    }
    const uint16_t* Bq = &Bs[0][0];
#endif
    __syncthreads();

    Frag af[2], bfr[4];
#pragma unroll
    for (int mi = 0; mi < 2; mi++) {
      int r = wm + mi * 16 + lrow;
      af[mi].u[0] = *(const uint4*)&As[r * 32 + lhalf * 8];
      af[mi].u[1] = *(const uint4*)&As[r * 32 + 16 + lhalf * 8];
    }
#pragma unroll
    for (int ni = 0; ni < 4; ni++) {
      int r = wn + ni * 16 + lrow;
      bfr[ni].u[0] = *(const uint4*)&Bq[r * 32 + lhalf * 8];
      bfr[ni].u[1] = *(const uint4*)&Bq[r * 32 + 16 + lhalf * 8];
    }
#pragma unroll
    for (int mi = 0; mi < 2; mi++)
#pragma unroll
      for (int ni = 0; ni < 4; ni++)
        acc[mi][ni] = __builtin_amdgcn_wmma_f32_16x16x32_bf16(
            false, af[mi].v, false, bfr[ni].v, (short)0, acc[mi][ni], false, false);
  }

#pragma unroll
  for (int mi = 0; mi < 2; mi++)
#pragma unroll
    for (int ni = 0; ni < 4; ni++) {
      int mb = m0 + wm + mi * 16 + 8 * lhalf;
      int n = n0 + wn + ni * 16 + lrow;
#pragma unroll
      for (int v = 0; v < 8; v++) {
        size_t idx = (size_t)(mb + v) * 1024 + n;
        out[idx] = acc[mi][ni][v] + resid[idx];
      }
    }
}

// ---------------------------------------------------------------------------
// launcher
// ---------------------------------------------------------------------------
extern "C" void kernel_launch(void* const* d_in, const int* in_sizes, int n_in,
                              void* d_out, int out_size, void* d_ws, size_t ws_size,
                              hipStream_t stream) {
  (void)in_sizes; (void)n_in; (void)out_size; (void)ws_size;
  const size_t ELT = 32ull * 1024 * 1024;   // elems per [B,S,D] tensor
  const int M = 32768;                      // B*S

  const float* h_in[2]    = {(const float*)d_in[0], (const float*)d_in[1]};
  const int*   mask_in[2] = {(const int*)d_in[2],   (const int*)d_in[3]};
  const float* W_in[2]    = {(const float*)d_in[4], (const float*)d_in[6]};
  const float* a_in[2]    = {(const float*)d_in[5], (const float*)d_in[7]};

  float* out = (float*)d_out;
  float* out_h[2]   = {out, out + ELT};
  float* out_att[2] = {out + 2 * ELT, out + 3 * ELT};

  // workspace carve-up
  uint8_t* w = (uint8_t*)d_ws;
  size_t off = 0;
  uint16_t* hb[2];  hb[0]  = (uint16_t*)(w + off); off += ELT * 2;
                    hb[1]  = (uint16_t*)(w + off); off += ELT * 2;
  uint16_t* WT[2];  WT[0]  = (uint16_t*)(w + off); off += 1048576ull * 2;
                    WT[1]  = (uint16_t*)(w + off); off += 1048576ull * 2;
  float* w1v[2];    w1v[0] = (float*)(w + off); off += 4096;
  float* w2v[2];    w2v[0] = (float*)(w + off); off += 4096;
                    w1v[1] = (float*)(w + off); off += 4096;
                    w2v[1] = (float*)(w + off); off += 4096;
  float* Wh1[2];    Wh1[0] = (float*)(w + off); off += (size_t)M * 4;
  float* Wh2[2];    Wh2[0] = (float*)(w + off); off += (size_t)M * 4;
                    Wh1[1] = (float*)(w + off); off += (size_t)M * 4;
                    Wh2[1] = (float*)(w + off); off += (size_t)M * 4;
  uint16_t* WhT[2]; WhT[0] = (uint16_t*)(w + off); off += ELT * 2;
                    WhT[1] = (uint16_t*)(w + off); off += ELT * 2;

  dim3 gemm_grid(8, 256), blk(256);

  for (int r = 0; r < 2; r++) {
    // h -> bf16
    cvt_f32_bf16_kernel<<<(int)(ELT / 4 / 256), blk, 0, stream>>>(
        h_in[r], hb[r], (int)(ELT / 4));
    // W -> W^T bf16
    wt_kernel<<<4096, blk, 0, stream>>>(W_in[r], WT[r]);
    // w1 = W a1, w2 = W a2
    wvec_kernel<<<1024, blk, 0, stream>>>(W_in[r], a_in[r], w1v[r], w2v[r]);
    // Wh1 = h w1, Wh2 = h w2
    wh12_kernel<<<M, blk, 0, stream>>>(hb[r], w1v[r], w2v[r], Wh1[r], Wh2[r]);
    // Wh^T = (h @ W)^T
    gemm1_bf16_kernel<<<gemm_grid, blk, 0, stream>>>(hb[r], WT[r], WhT[r]);
    // att
    softmax_att_kernel<<<M, blk, 0, stream>>>(Wh1[r], Wh2[r], mask_in[r], out_att[r]);
    // out = att @ Wh + h
    gemm2_bf16_kernel<<<gemm_grid, blk, 0, stream>>>(out_att[r], WhT[r], h_in[r], out_h[r]);
  }
}
